// TransMIL_32976758899136
// MI455X (gfx1250) — matrix operand order repaired and checked by reading, not compiled, not run on hardware
//
#include <hip/hip_runtime.h>
#include <hip/hip_bf16.h>

// ---------------------------------------------------------------------------
// TransMIL forward for MI455X (gfx1250, wave32, WMMA).
// All GEMMs run through one generic bf16-WMMA kernel (v_wmma_f32_16x16x32_bf16,
// f32 accumulate). f32 data in HBM, bf16 only inside LDS/fragments.
// ---------------------------------------------------------------------------

typedef __attribute__((ext_vector_type(16))) __bf16 v16bf;
typedef __attribute__((ext_vector_type(8)))  __bf16 v8bf;
typedef __attribute__((ext_vector_type(4)))  __bf16 v4bf;
typedef __attribute__((ext_vector_type(8)))  float  v8f;
typedef __attribute__((ext_vector_type(4)))  float  f4v;   // primitive float4

// explicit global-address-space access (forces global_load_*/global_store_*,
// keeping LOADcnt/STOREcnt decoupled from DScnt, unlike flat_*).
// NOTE: must be primitive types (ext_vector / scalar), not HIP_vector_type
// classes, so no constructors are involved in the AS(1) dereference.
typedef __attribute__((address_space(1))) float gf1;
typedef __attribute__((address_space(1))) f4v   gf4;

#define DEV __device__ __forceinline__

DEV f4v gload4(const float* p) {
  return *(const gf4*)(unsigned long long)p;
}
DEV float gloadf(const float* p) {
  return *(const gf1*)(unsigned long long)p;
}
DEV void gstoref(float* p, float v) {
  *(gf1*)(unsigned long long)p = v;
}

// model dims
constexpr int BATCH = 2;
constexpr int NTOK  = 10001;   // cls + 100*100 tokens
constexpr int NP    = 10240;   // nystrom-padded sequence (pad 239 at front)
constexpr int PADF  = 239;
constexpr int DIM   = 512;
constexpr int NH    = 8;
constexpr int DH    = 64;
constexpr int LMK   = 256;     // landmarks
constexpr int LSEG  = 40;      // NP / LMK
constexpr float QSCALE = 0.125f; // 64^-0.5

// ---------------------------------------------------------------------------
// block reductions (wave32)
// ---------------------------------------------------------------------------
DEV float wave_red_sum(float v) {
#pragma unroll
  for (int m = 16; m; m >>= 1) v += __shfl_xor(v, m, 32);
  return v;
}
DEV float wave_red_max(float v) {
#pragma unroll
  for (int m = 16; m; m >>= 1) v = fmaxf(v, __shfl_xor(v, m, 32));
  return v;
}
__device__ float block_sum(float v) {
  __shared__ float red[8];
  int lane = threadIdx.x & 31, w = threadIdx.x >> 5;
  int nw = blockDim.x >> 5;
  v = wave_red_sum(v);
  __syncthreads();
  if (lane == 0) red[w] = v;
  __syncthreads();
  float s = 0.f;
  for (int i = 0; i < nw; ++i) s += red[i];
  return s;
}
__device__ float block_max(float v) {
  __shared__ float redm[8];
  int lane = threadIdx.x & 31, w = threadIdx.x >> 5;
  int nw = blockDim.x >> 5;
  v = wave_red_max(v);
  __syncthreads();
  if (lane == 0) redm[w] = v;
  __syncthreads();
  float s = -3.4e38f;
  for (int i = 0; i < nw; ++i) s = fmaxf(s, redm[i]);
  return s;
}

DEV v4bf pack4(f4v v) {
  v4bf p;
  p[0] = (__bf16)v[0]; p[1] = (__bf16)v[1];
  p[2] = (__bf16)v[2]; p[3] = (__bf16)v[3];
  return p;
}

// ---------------------------------------------------------------------------
// Generic batched bf16 WMMA GEMM:  C = diag_c*I + alpha*(A @ opB(B)) [+bias][+C][relu]
//   A: row-major MxK (lda), B: KxN row-major (tB=0) or NxK row-major (tB=1)
//   per-batch-z operand offset = (z/hdiv)*sXo + (z%hdiv)*sXi  (for (b,h) batching)
// Tile: 128x64, BK=32, 256 threads = 8 waves, each wave 32x32 (2x2 WMMA tiles).
// Staging: unconditional clamped global_load_b128 (OOB rows read row 0, value
// zeroed by cndmask) batched ahead of packed-bf16 LDS stores.
// ---------------------------------------------------------------------------
#define BM 128
#define BN 64
#define BK 32
#define ASTR 40   // bf16 elems per A LDS row (80B, 16B aligned, conflict-free)
#define BSTR 72   // bf16 elems per B LDS row (144B, 16B aligned)

#define GF_RELU  1
#define GF_ACCUM 2

__global__ __launch_bounds__(256) void gemm_bf16_wmma(
    const float* __restrict__ A, long long sAo, long long sAi, int lda,
    const float* __restrict__ Bm, long long sBo, long long sBi, int ldb, int tB,
    float* __restrict__ C, long long sCo, long long sCi, int ldc,
    const float* __restrict__ bias,
    int M, int N, int K, int hdiv, float alpha, float diagc, int flags)
{
  __shared__ alignas(16) __bf16 As[BM * ASTR];
  __shared__ alignas(16) __bf16 Bs[BK * BSTR];

  const int z  = blockIdx.z;
  const int zo = z / hdiv, zi = z % hdiv;
  A  += zo * sAo + zi * sAi;
  Bm += zo * sBo + zi * sBi;
  C  += zo * sCo + zi * sCi;

  const int m0 = blockIdx.y * BM;
  const int n0 = blockIdx.x * BN;
  const int t = threadIdx.x;
  const int lane = t & 31, wid = t >> 5;
  const int lo = lane & 15, hi = lane >> 4;
  const int wm = (wid & 3) * 32;   // wave M offset in tile
  const int wn = (wid >> 2) * 32;  // wave N offset in tile
  const bool fullM = (m0 + BM) <= M;

  // ---- per-thread staging pointers (computed once) ----
  // A: thread -> (row = t>>3 (+32*i), col4 = (t&7)*4), 4 float4 per K-step
  const int arow = t >> 3;
  const int acol = (t & 7) * 4;
  const float* aptr[4];
  bool aok[4];
  int asm_[4];
#pragma unroll
  for (int i = 0; i < 4; ++i) {
    int r = arow + 32 * i;
    int gr = m0 + r;
    aok[i] = (gr < M);
    aptr[i] = A + (long long)(aok[i] ? gr : 0) * lda + acol;  // clamped, always valid
    asm_[i] = r * ASTR + acol;
  }
  // B: 2 float4 per K-step
  const float* bptr[2];
  int bsm[2];
  long long bstep;
  if (!tB) {                       // B is KxN: (row = t>>4 (+16*j), col4 = (t&15)*4)
    int brow = t >> 4;
    int bcol = (t & 15) * 4;
#pragma unroll
    for (int j = 0; j < 2; ++j) {
      bptr[j] = Bm + (long long)(brow + 16 * j) * ldb + (n0 + bcol);
      bsm[j] = (brow + 16 * j) * BSTR + bcol;
    }
    bstep = (long long)BK * ldb;
  } else {                         // B is NxK: (col = t&63, krow4 = (t>>6)*4 (+16*j))
    int c = t & 63;
    int kch = (t >> 6) * 4;
#pragma unroll
    for (int j = 0; j < 2; ++j) {
      bptr[j] = Bm + (long long)(n0 + c) * ldb + (kch + 16 * j);
      bsm[j] = (kch + 16 * j) * BSTR + c;
    }
    bstep = (long long)BK;
  }

  v8f acc[2][2];
#pragma unroll
  for (int i = 0; i < 2; ++i)
#pragma unroll
    for (int j = 0; j < 2; ++j)
#pragma unroll
      for (int e = 0; e < 8; ++e) acc[i][j][e] = 0.f;

  const f4v zero4 = {0.f, 0.f, 0.f, 0.f};

  for (int k0 = 0; k0 < K; k0 += BK) {
    // speculative prefetch of next K-tile (dropped safely if invalid)
    if (k0 + BK < K) {
      __builtin_prefetch(aptr[0] + BK, 0, 3);
      __builtin_prefetch(bptr[0] + bstep, 0, 3);
    }
    // issue all 6 global_load_b128 back-to-back (no exec predication)
    f4v av[4], bv[2];
#pragma unroll
    for (int i = 0; i < 4; ++i) { av[i] = gload4(aptr[i]); aptr[i] += BK; }
#pragma unroll
    for (int j = 0; j < 2; ++j) { bv[j] = gload4(bptr[j]); bptr[j] += bstep; }
    // convert + store to LDS (zero OOB rows via cndmask, not exec mask)
#pragma unroll
    for (int i = 0; i < 4; ++i) {
      f4v v = aok[i] ? av[i] : zero4;
      *(v4bf*)&As[asm_[i]] = pack4(v);
    }
    if (!tB) {
#pragma unroll
      for (int j = 0; j < 2; ++j) *(v4bf*)&Bs[bsm[j]] = pack4(bv[j]);
    } else {
#pragma unroll
      for (int j = 0; j < 2; ++j) {
        Bs[bsm[j] + 0 * BSTR] = (__bf16)bv[j][0];
        Bs[bsm[j] + 1 * BSTR] = (__bf16)bv[j][1];
        Bs[bsm[j] + 2 * BSTR] = (__bf16)bv[j][2];
        Bs[bsm[j] + 3 * BSTR] = (__bf16)bv[j][3];
      }
    }
    __syncthreads();

    // fragments per ISA 7.12.2 (16-bit A 16x32; B lane=N col, K in same order as A)
    const int K1 = hi * 8, K2 = 16 + hi * 8;
    v16bf afr[2], bfr[2];
#pragma unroll
    for (int ti = 0; ti < 2; ++ti) {
      const __bf16* ap = &As[(wm + ti * 16 + lo) * ASTR];
      v8bf x0 = *(const v8bf*)(ap + K1);
      v8bf x1 = *(const v8bf*)(ap + K2);
      afr[ti] = __builtin_shufflevector(x0, x1, 0,1,2,3,4,5,6,7,8,9,10,11,12,13,14,15);
    }
#pragma unroll
    for (int tj = 0; tj < 2; ++tj) {
      const __bf16* bp = &Bs[wn + tj * 16 + lo];
#pragma unroll
      for (int s = 0; s < 8; ++s) {
        bfr[tj][s] = bp[(K1 + s) * BSTR];
        bfr[tj][s + 8] = bp[(K2 + s) * BSTR];
      }
    }
#pragma unroll
    for (int ti = 0; ti < 2; ++ti)
#pragma unroll
      for (int tj = 0; tj < 2; ++tj)
        acc[ti][tj] = __builtin_amdgcn_wmma_f32_16x16x32_bf16(
            false, afr[ti], false, bfr[tj], (short)0, acc[ti][tj], false, false);
    __syncthreads();
  }

  // epilogue: D layout -> row = r + 8*hi, col = lo (per 16x16 f32 C/D table)
#pragma unroll
  for (int ti = 0; ti < 2; ++ti)
#pragma unroll
    for (int tj = 0; tj < 2; ++tj)
#pragma unroll
      for (int r = 0; r < 8; ++r) {
        int gm = m0 + wm + ti * 16 + r + 8 * hi;
        int gn = n0 + wn + tj * 16 + lo;
        if (fullM || gm < M) {
          float v = alpha * acc[ti][tj][r];
          if (gm == gn) v += diagc;
          if (bias) v += gloadf(&bias[gn]);
          float* cp = &C[(long long)gm * ldc + gn];
          if (flags & GF_ACCUM) v += gloadf(cp);
          if (flags & GF_RELU) v = fmaxf(v, 0.f);
          gstoref(cp, v);
        }
      }
}

// ---------------------------------------------------------------------------
// small kernels
// ---------------------------------------------------------------------------

// LayerNorm(x) written into front-zero-padded xp [B][NP][DIM]
__global__ __launch_bounds__(128) void k_ln_pad(const float* __restrict__ x,
    const float* __restrict__ g, const float* __restrict__ bta,
    float* __restrict__ xp)
{
  int row = blockIdx.x, b = blockIdx.y, t = threadIdx.x;
  float* dst = xp + ((long long)b * NP + row) * DIM;
  if (row < PADF) {
#pragma unroll
    for (int i = 0; i < 4; ++i) gstoref(&dst[t + i * 128], 0.f);
    return;
  }
  const float* src = x + ((long long)b * NTOK + (row - PADF)) * DIM;
  float v[4]; float s = 0.f;
#pragma unroll
  for (int i = 0; i < 4; ++i) { v[i] = gloadf(&src[t + 128 * i]); s += v[i]; }
  float mu = block_sum(s) * (1.f / 512.f);
  float q = 0.f;
#pragma unroll
  for (int i = 0; i < 4; ++i) { float d = v[i] - mu; q += d * d; }
  float var = block_sum(q) * (1.f / 512.f);
  float rs = rsqrtf(var + 1e-5f);
#pragma unroll
  for (int i = 0; i < 4; ++i) {
    int c = t + 128 * i;
    gstoref(&dst[c], (v[i] - mu) * rs * gloadf(&g[c]) + gloadf(&bta[c]));
  }
}

// landmark means: ql/kl [bh][256][64] = mean over 40-token segments
__global__ __launch_bounds__(256) void k_landmark(const float* __restrict__ qkv,
    float* __restrict__ ql, float* __restrict__ kl)
{
  int idx = blockIdx.x * 256 + threadIdx.x;           // 16*256*64
  int d = idx & 63, m = (idx >> 6) & 255, bh = idx >> 14;
  int b = bh >> 3, h = bh & 7;
  const float* qb = qkv + (long long)b * NP * 1536 + (long long)m * LSEG * 1536 + h * 64 + d;
  float sq = 0.f, sk = 0.f;
  for (int j = 0; j < LSEG; ++j) {
    sq += gloadf(&qb[(long long)j * 1536]);
    sk += gloadf(&qb[(long long)j * 1536 + DIM]);
  }
  gstoref(&ql[idx], sq * (1.f / LSEG));
  gstoref(&kl[idx], sk * (1.f / LSEG));
}

// in-place row softmax, row length RL, one block per row
__global__ __launch_bounds__(256) void k_softmax(float* __restrict__ s, int RL)
{
  float* p = s + (long long)blockIdx.x * RL;
  int t = threadIdx.x;
  float mx = -3.4e38f;
  for (int c = t; c < RL; c += 256) mx = fmaxf(mx, gloadf(&p[c]));
  mx = block_max(mx);
  float sum = 0.f;
  for (int c = t; c < RL; c += 256) {
    float e = __expf(gloadf(&p[c]) - mx);
    gstoref(&p[c], e);
    sum += e;
  }
  sum = block_sum(sum);
  float inv = 1.f / sum;
  for (int c = t; c < RL; c += 256) gstoref(&p[c], gloadf(&p[c]) * inv);
}

__global__ void k_zero2(float* s) { if (threadIdx.x == 0) { s[0] = 0.f; s[1] = 0.f; } }

// global max row-abs-sum and col-abs-sum of a2 (over all bh)
__global__ __launch_bounds__(256) void k_a2norm(const float* __restrict__ a2, float* scal)
{
  int bh = blockIdx.x, t = threadIdx.x;
  const float* p = a2 + (long long)bh * LMK * LMK;
  float rs = 0.f, cs = 0.f;
  for (int j = 0; j < LMK; ++j) {
    rs += fabsf(gloadf(&p[(long long)t * LMK + j]));
    cs += fabsf(gloadf(&p[(long long)j * LMK + t]));
  }
  float rm = block_max(rs);
  float cm = block_max(cs);
  if (t == 0) {
    atomicMax((int*)&scal[0], __float_as_int(rm));
    atomicMax((int*)&scal[1], __float_as_int(cm));
  }
}

// z0 = a2^T / (max_rowsum * max_colsum)
__global__ __launch_bounds__(256) void k_zinit(const float* __restrict__ a2,
    const float* __restrict__ scal, float* __restrict__ z)
{
  long long idx = (long long)blockIdx.x * 256 + threadIdx.x;  // 16*256*256
  int j = (int)(idx & 255), i = (int)((idx >> 8) & 255);
  long long bh = idx >> 16;
  float inv = 1.f / (gloadf(&scal[0]) * gloadf(&scal[1]));
  gstoref(&z[idx], gloadf(&a2[(bh << 16) + (long long)j * LMK + i]) * inv);
}

// Y = c*I - X  (16 batches of 256x256)
__global__ __launch_bounds__(256) void k_diagsub(const float* __restrict__ X,
    float* __restrict__ Y, float c)
{
  long long idx = (long long)blockIdx.x * 256 + threadIdx.x;
  int j = (int)(idx & 255), i = (int)((idx >> 8) & 255);
  gstoref(&Y[idx], ((i == j) ? c : 0.f) - gloadf(&X[idx]));
}

// depthwise 33-tap conv along sequence on v, accumulated into ctx[b][n][h*64+d]
__global__ __launch_bounds__(256) void k_resconv(const float* __restrict__ qkv,
    const float* __restrict__ w, float* __restrict__ ctx)
{
  long long idx = (long long)blockIdx.x * 256 + threadIdx.x;  // B*NP*DIM
  int c = (int)(idx % 512); long long r = idx / 512;
  int n = (int)(r % NP); int b = (int)(r / NP);
  int h = c >> 6, d = c & 63;
  const float* vb = qkv + (long long)b * NP * 1536 + 2 * DIM + h * 64 + d;
  const float* wp = w + h * 33;
  float acc = 0.f;
#pragma unroll
  for (int k = 0; k < 33; ++k) {
    int nn = n + k - 16;
    if ((unsigned)nn < (unsigned)NP) acc += gloadf(&wp[k]) * gloadf(&vb[(long long)nn * 1536]);
  }
  gstoref(&ctx[idx], gloadf(&ctx[idx]) + acc);
}

// PPEG: y = x + dw7 + b7 + dw5 + b5 + dw3 + b3 on 100x100 grid (tokens 1..10000)
__global__ __launch_bounds__(256) void k_ppeg(const float* __restrict__ x,
    const float* __restrict__ w7, const float* __restrict__ b7,
    const float* __restrict__ w5, const float* __restrict__ b5,
    const float* __restrict__ w3, const float* __restrict__ b3,
    float* __restrict__ y)
{
  long long idx = (long long)blockIdx.x * 256 + threadIdx.x;  // B*10000*DIM
  int c = (int)(idx % 512); long long r = idx / 512;
  int n = (int)(r % 10000); int b = (int)(r / 10000);
  int rr = n / 100, cc = n % 100;
  const float* xb = x + ((long long)b * NTOK + 1) * DIM + c;
  float acc = gloadf(&xb[(long long)n * DIM]);
  const float* wp7 = w7 + c * 49;
#pragma unroll
  for (int i = 0; i < 7; ++i) { int yy = rr + i - 3; if ((unsigned)yy < 100u) {
#pragma unroll
    for (int j = 0; j < 7; ++j) { int xx = cc + j - 3; if ((unsigned)xx < 100u)
      acc += gloadf(&wp7[i * 7 + j]) * gloadf(&xb[(long long)(yy * 100 + xx) * DIM]); } } }
  acc += gloadf(&b7[c]);
  const float* wp5 = w5 + c * 25;
#pragma unroll
  for (int i = 0; i < 5; ++i) { int yy = rr + i - 2; if ((unsigned)yy < 100u) {
#pragma unroll
    for (int j = 0; j < 5; ++j) { int xx = cc + j - 2; if ((unsigned)xx < 100u)
      acc += gloadf(&wp5[i * 5 + j]) * gloadf(&xb[(long long)(yy * 100 + xx) * DIM]); } } }
  acc += gloadf(&b5[c]);
  const float* wp3 = w3 + c * 9;
#pragma unroll
  for (int i = 0; i < 3; ++i) { int yy = rr + i - 1; if ((unsigned)yy < 100u) {
#pragma unroll
    for (int j = 0; j < 3; ++j) { int xx = cc + j - 1; if ((unsigned)xx < 100u)
      acc += gloadf(&wp3[i * 3 + j]) * gloadf(&xb[(long long)(yy * 100 + xx) * DIM]); } } }
  acc += gloadf(&b3[c]);
  gstoref(&y[((long long)b * NTOK + 1 + n) * DIM + c], acc);
}

__global__ void k_cls(const float* __restrict__ cls, float* __restrict__ x)
{
  int t = blockIdx.x * 256 + threadIdx.x;   // 1024
  int b = t >> 9, c = t & 511;
  gstoref(&x[(long long)b * NTOK * DIM + c], gloadf(&cls[c]));
}

__global__ void k_row0copy(const float* __restrict__ src, float* __restrict__ dst)
{
  int t = blockIdx.x * 256 + threadIdx.x;   // 1024
  int b = t >> 9, c = t & 511;
  gstoref(&dst[(long long)b * NTOK * DIM + c], gloadf(&src[(long long)b * NTOK * DIM + c]));
}

// final LN on cls token + fc2 head
__global__ __launch_bounds__(256) void k_head(const float* __restrict__ x,
    const float* __restrict__ g, const float* __restrict__ bta,
    const float* __restrict__ w2, const float* __restrict__ b2,
    float* __restrict__ out)
{
  int b = blockIdx.x, t = threadIdx.x;
  const float* p = x + (long long)b * NTOK * DIM;
  float v0 = gloadf(&p[t]), v1 = gloadf(&p[t + 256]);
  float mu = block_sum(v0 + v1) * (1.f / 512.f);
  float d0 = v0 - mu, d1 = v1 - mu;
  float var = block_sum(d0 * d0 + d1 * d1) * (1.f / 512.f);
  float rs = rsqrtf(var + 1e-5f);
  float z0 = d0 * rs * gloadf(&g[t]) + gloadf(&bta[t]);
  float z1 = d1 * rs * gloadf(&g[t + 256]) + gloadf(&bta[t + 256]);
  float l0 = z0 * gloadf(&w2[t * 2]) + z1 * gloadf(&w2[(t + 256) * 2]);
  float l1 = z0 * gloadf(&w2[t * 2 + 1]) + z1 * gloadf(&w2[(t + 256) * 2 + 1]);
  l0 = block_sum(l0);
  l1 = block_sum(l1);
  if (t == 0) {
    gstoref(&out[b * 2], l0 + gloadf(&b2[0]));
    gstoref(&out[b * 2 + 1], l1 + gloadf(&b2[1]));
  }
}

// ---------------------------------------------------------------------------
// host orchestration
// ---------------------------------------------------------------------------
static inline void launch_gemm(hipStream_t st,
    const float* A, long long sAo, long long sAi, int lda,
    const float* Bm, long long sBo, long long sBi, int ldb, int tB,
    float* C, long long sCo, long long sCi, int ldc, const float* bias,
    int M, int N, int K, int bat, int hdiv, float alpha, float diagc, int flags)
{
  dim3 g(N / BN, (M + BM - 1) / BM, bat);
  gemm_bf16_wmma<<<g, 256, 0, st>>>(A, sAo, sAi, lda, Bm, sBo, sBi, ldb, tB,
                                    C, sCo, sCi, ldc, bias, M, N, K, hdiv,
                                    alpha, diagc, flags);
}

struct Scratch {
  float *xp, *qkv, *ql, *kl, *s1, *s2, *s3, *zA, *zB, *xz, *P1, *P2, *a3v, *tt, *ctx, *scal;
};

static void trans_layer(hipStream_t st, float* X,
    const float* ng, const float* nb, const float* qkvw,
    const float* ow, const float* ob, const float* rw, const Scratch& W)
{
  const long long SQL = (long long)LMK * DH;     // 16384
  const long long SA2 = (long long)LMK * LMK;    // 65536
  // 1. LN + front pad
  k_ln_pad<<<dim3(NP, BATCH), 128, 0, st>>>(X, ng, nb, W.xp);
  // 2. qkv projection (no bias)
  launch_gemm(st, W.xp, (long long)NP * DIM, 0, DIM, qkvw, 0, 0, 1536, 0,
              W.qkv, (long long)NP * 1536, 0, 1536, nullptr,
              NP, 1536, DIM, BATCH, 1, 1.f, 0.f, 0);
  // 3. landmarks
  k_landmark<<<(16 * LMK * DH) / 256, 256, 0, st>>>(W.qkv, W.ql, W.kl);
  // 4. a1 = softmax(scale * q @ kl^T)    [bh][NP][LMK]
  launch_gemm(st, W.qkv, (long long)NP * 1536, 64, 1536,
              W.kl, 8 * SQL, SQL, DH, 1,
              W.s1, 8LL * NP * LMK, (long long)NP * LMK, LMK, nullptr,
              NP, LMK, DH, 16, 8, QSCALE, 0.f, 0);
  k_softmax<<<16 * NP, 256, 0, st>>>(W.s1, LMK);
  // 5. a2 = softmax(scale * ql @ kl^T)   [bh][LMK][LMK]
  launch_gemm(st, W.ql, 8 * SQL, SQL, DH, W.kl, 8 * SQL, SQL, DH, 1,
              W.s2, 8 * SA2, SA2, LMK, nullptr,
              LMK, LMK, DH, 16, 8, QSCALE, 0.f, 0);
  k_softmax<<<16 * LMK, 256, 0, st>>>(W.s2, LMK);
  // 6. a3 = softmax(scale * ql @ k^T)    [bh][LMK][NP]
  launch_gemm(st, W.ql, 8 * SQL, SQL, DH,
              W.qkv + DIM, (long long)NP * 1536, 64, 1536, 1,
              W.s3, 8LL * LMK * NP, (long long)LMK * NP, NP, nullptr,
              LMK, NP, DH, 16, 8, QSCALE, 0.f, 0);
  k_softmax<<<16 * LMK, 256, 0, st>>>(W.s3, NP);
  // 7. Newton-Schulz pinv of a2 (6 iterations)
  k_zero2<<<1, 32, 0, st>>>(W.scal);
  k_a2norm<<<16, 256, 0, st>>>(W.s2, W.scal);
  k_zinit<<<(16 * LMK * LMK) / 256, 256, 0, st>>>(W.s2, W.scal, W.zA);
  float* zc = W.zA; float* zn = W.zB;
  for (int it = 0; it < 6; ++it) {
    launch_gemm(st, W.s2, SA2, 0, LMK, zc, SA2, 0, LMK, 0,
                W.xz, SA2, 0, LMK, nullptr, LMK, LMK, LMK, 16, 1, 1.f, 0.f, 0);
    k_diagsub<<<(16 * LMK * LMK) / 256, 256, 0, st>>>(W.xz, W.P1, 7.f);
    launch_gemm(st, W.xz, SA2, 0, LMK, W.P1, SA2, 0, LMK, 0,
                W.P2, SA2, 0, LMK, nullptr, LMK, LMK, LMK, 16, 1, -1.f, 15.f, 0);
    launch_gemm(st, W.xz, SA2, 0, LMK, W.P2, SA2, 0, LMK, 0,
                W.P1, SA2, 0, LMK, nullptr, LMK, LMK, LMK, 16, 1, -1.f, 13.f, 0);
    launch_gemm(st, zc, SA2, 0, LMK, W.P1, SA2, 0, LMK, 0,
                zn, SA2, 0, LMK, nullptr, LMK, LMK, LMK, 16, 1, 0.25f, 0.f, 0);
    float* tmp = zc; zc = zn; zn = tmp;
  }
  // 8. a3v = a3 @ v                       [bh][LMK][DH]
  launch_gemm(st, W.s3, 8LL * LMK * NP, (long long)LMK * NP, NP,
              W.qkv + 2 * DIM, (long long)NP * 1536, 64, 1536, 0,
              W.a3v, 8 * SQL, SQL, DH, nullptr,
              LMK, DH, NP, 16, 8, 1.f, 0.f, 0);
  // 9. tt = pinv(a2) @ a3v
  launch_gemm(st, zc, SA2, 0, LMK, W.a3v, SQL, 0, DH, 0,
              W.tt, SQL, 0, DH, nullptr, LMK, DH, LMK, 16, 1, 1.f, 0.f, 0);
  // 10. ctx = a1 @ tt, scattered by head into [b][n][h*64]
  launch_gemm(st, W.s1, 8LL * NP * LMK, (long long)NP * LMK, LMK,
              W.tt, 8 * SQL, SQL, DH, 0,
              W.ctx, (long long)NP * DIM, DH, DIM, nullptr,
              NP, DH, LMK, 16, 8, 1.f, 0.f, 0);
  // 11. depthwise residual conv on v
  k_resconv<<<(BATCH * NP * DIM) / 256, 256, 0, st>>>(W.qkv, rw, W.ctx);
  // 12. out proj + bias, slice last NTOK rows, accumulate into residual X
  launch_gemm(st, W.ctx + (long long)PADF * DIM, (long long)NP * DIM, 0, DIM,
              ow, 0, 0, DIM, 0,
              X, (long long)NTOK * DIM, 0, DIM, ob,
              NTOK, DIM, DIM, BATCH, 1, 1.f, 0.f, GF_ACCUM);
}

extern "C" void kernel_launch(void* const* d_in, const int* in_sizes, int n_in,
                              void* d_out, int out_size, void* d_ws, size_t ws_size,
                              hipStream_t stream)
{
  (void)in_sizes; (void)n_in; (void)out_size; (void)ws_size;
  const float* h      = (const float*)d_in[0];
  const float* fc1_w  = (const float*)d_in[1];
  const float* fc1_b  = (const float*)d_in[2];
  const float* cls    = (const float*)d_in[3];
  const float* l1_ng  = (const float*)d_in[4];
  const float* l1_nb  = (const float*)d_in[5];
  const float* l1_qkv = (const float*)d_in[6];
  const float* l1_ow  = (const float*)d_in[7];
  const float* l1_ob  = (const float*)d_in[8];
  const float* l1_rw  = (const float*)d_in[9];
  const float* l2_ng  = (const float*)d_in[10];
  const float* l2_nb  = (const float*)d_in[11];
  const float* l2_qkv = (const float*)d_in[12];
  const float* l2_ow  = (const float*)d_in[13];
  const float* l2_ob  = (const float*)d_in[14];
  const float* l2_rw  = (const float*)d_in[15];
  const float* p7w = (const float*)d_in[16];
  const float* p7b = (const float*)d_in[17];
  const float* p5w = (const float*)d_in[18];
  const float* p5b = (const float*)d_in[19];
  const float* p3w = (const float*)d_in[20];
  const float* p3b = (const float*)d_in[21];
  const float* n_g = (const float*)d_in[22];
  const float* n_b = (const float*)d_in[23];
  const float* fc2w = (const float*)d_in[24];
  const float* fc2b = (const float*)d_in[25];
  float* out = (float*)d_out;

  // workspace carve-out (~660 MB total)
  char* base = (char*)d_ws;
  size_t off = 0;
  auto alloc = [&](size_t nfloats) -> float* {
    float* p = (float*)(base + off);
    off += (nfloats * sizeof(float) + 255) & ~(size_t)255;
    return p;
  };
  float* xA = alloc((size_t)BATCH * NTOK * DIM);
  float* xB = alloc((size_t)BATCH * NTOK * DIM);
  Scratch W;
  W.xp   = alloc((size_t)BATCH * NP * DIM);
  W.qkv  = alloc((size_t)BATCH * NP * 1536);
  W.ql   = alloc((size_t)16 * LMK * DH);
  W.kl   = alloc((size_t)16 * LMK * DH);
  W.s1   = alloc((size_t)16 * NP * LMK);
  W.s3   = alloc((size_t)16 * LMK * NP);
  W.s2   = alloc((size_t)16 * LMK * LMK);
  W.zA   = alloc((size_t)16 * LMK * LMK);
  W.zB   = alloc((size_t)16 * LMK * LMK);
  W.xz   = alloc((size_t)16 * LMK * LMK);
  W.P1   = alloc((size_t)16 * LMK * LMK);
  W.P2   = alloc((size_t)16 * LMK * LMK);
  W.a3v  = alloc((size_t)16 * LMK * DH);
  W.tt   = alloc((size_t)16 * LMK * DH);
  W.ctx  = alloc((size_t)BATCH * NP * DIM);
  W.scal = alloc(8);

  // fc1 + relu -> xA rows 1..10000 ; cls token -> row 0
  launch_gemm(stream, h, 10000LL * 1024, 0, 1024, fc1_w, 0, 0, DIM, 0,
              xA + DIM, (long long)NTOK * DIM, 0, DIM, fc1_b,
              10000, DIM, 1024, BATCH, 1, 1.f, 0.f, GF_RELU);
  k_cls<<<4, 256, 0, stream>>>(cls, xA);

  // transformer layer 1 (in-place residual on xA)
  trans_layer(stream, xA, l1_ng, l1_nb, l1_qkv, l1_ow, l1_ob, l1_rw, W);

  // PPEG: xA -> xB
  k_ppeg<<<(BATCH * 10000 * DIM) / 256, 256, 0, stream>>>(
      xA, p7w, p7b, p5w, p5b, p3w, p3b, xB);
  k_row0copy<<<4, 256, 0, stream>>>(xA, xB);

  // transformer layer 2 (in-place on xB)
  trans_layer(stream, xB, l2_ng, l2_nb, l2_qkv, l2_ow, l2_ob, l2_rw, W);

  // final LN on cls token + fc2
  k_head<<<BATCH, 256, 0, stream>>>(xB, n_g, n_b, fc2w, fc2b, out);
}
